// SelfAttn_65876208386480
// MI455X (gfx1250) — matrix operand order repaired and checked by reading, not compiled, not run on hardware
//
#include <hip/hip_runtime.h>
#include <stdint.h>

// ---------------------------------------------------------------------------
// Self-attention (B=8, E=512, DIN=64, H=15, DQK=DV=512) for MI455X / gfx1250.
//
// Algebraic refactor (exact in real arithmetic):
//   score = G*(W1q W1k^T/sqrt(d))*G^T + (G p)1^T + 1(G q)^T + s + bias
//   out   = sum_h (attn_h * G) * (Wv_h Wo_h) + (bv Wo + bo)
// with 64x64 per-head matrices M_h, C_h  ->  ~10 GF of matmuls instead of
// ~80 GF; run is bound by the mandatory 126 MB attn write (~5.4 us @ 23.3TB/s).
//
// CDNA5 specifics used:
//  * v_wmma_f32_16x16x32_bf16 (wave32 WMMA, fp32 accumulate), dual-tile
//    accumulator chains to fill the 4-slot WMMA->VALU co-exec hazard window
//  * global_load_async_to_lds_b128 + s_wait_asynccnt for LDS staging
//  * non-temporal stores for the streaming attn output (keeps G_bias in L2)
//  * global_prefetch for the per-WG bias tile
//  * 283 KB of the WGP's 320 KB LDS, 8 waves/WG (2 per SIMD32)
// ---------------------------------------------------------------------------

#define H_    15
#define B_    8
#define E_    512
#define DIN_  64
#define DQK_  512
#define DV_   512
#define DOUT_ 64
#define NEGV  (-1.0e9f)
#define RS_   0.044194173824159216f   // 1/sqrt(512)

typedef __bf16 bf16_t;
typedef bf16_t bf16x16 __attribute__((ext_vector_type(16)));
typedef bf16_t bf16x8  __attribute__((ext_vector_type(8)));
typedef float  f32x8   __attribute__((ext_vector_type(8)));

// ---- workspace byte offsets ----------------------------------------------
#define WSB_MTBF 0u         // bf16 [15][64][64]  MT[h][j][i]
#define WSB_CTBF 122880u    // bf16 [15][64][64]  CT[h][o][j]
#define WSB_PV   245760u    // f32  [15][64]
#define WSB_QV   249600u    // f32  [15][64]
#define WSB_SV   253440u    // f32  [15]
#define WSB_DV   253504u    // f32  [64]
#define WSB_COL  253952u    // f32  [8][15][512]  colterm (incl. s_h)
#define WSB_ROW  499712u    // f32  [8][15][512]  rowterm
#define WSB_GBF  745472u    // bf16 [8][512][64]
#define WSB_GBFT 1269760u   // bf16 [8][64][512]
#define WSB_ACC  1794048u   // f32  [15][8*512*64] per-head out contributions
// total 17522688 bytes (~16.7 MB)

// ---- WMMA fragment helpers (wave32 layouts per CDNA5 ISA 7.12.2) ----------
__device__ __forceinline__ bf16x16 load_afrag(const bf16_t* base, int stride,
                                              int m0, int k0, int lane) {
  const bf16_t* p = base + (m0 + (lane & 15)) * stride + k0 + ((lane >> 4) << 3);
  bf16x8 lo = *(const bf16x8*)p;
  bf16x8 hi = *(const bf16x8*)(p + 16);
  return __builtin_shufflevector(lo, hi, 0,1,2,3,4,5,6,7,8,9,10,11,12,13,14,15);
}
__device__ __forceinline__ bf16x16 load_bfrag(const bf16_t* baseT, int stride,
                                              int n0, int k0, int lane) {
  const bf16_t* p = baseT + (n0 + (lane & 15)) * stride + k0 + ((lane >> 4) << 3);
  bf16x8 lo = *(const bf16x8*)p;
  bf16x8 hi = *(const bf16x8*)(p + 16);
  return __builtin_shufflevector(lo, hi, 0,1,2,3,4,5,6,7,8,9,10,11,12,13,14,15);
}
__device__ __forceinline__ f32x8 zero8() {
  f32x8 z;
#pragma unroll
  for (int i = 0; i < 8; ++i) z[i] = 0.0f;
  return z;
}
__device__ __forceinline__ f32x8 wmma_bf16(bf16x16 a, bf16x16 b, f32x8 c) {
  return __builtin_amdgcn_wmma_f32_16x16x32_bf16(false, a, false, b,
                                                 (short)0, c, false, false);
}

// ---- CDNA5 async global->LDS copy (tracked by ASYNCcnt) -------------------
// Each enabled lane copies 16 bytes: LDS[vdst] = MEM[vaddr64].
__device__ __forceinline__ void lds_copy_async(uint32_t lds_base,
                                               const char* src, int bytes,
                                               int tid) {
  for (int off = tid * 16; off < bytes; off += 256 * 16) {
    asm volatile("global_load_async_to_lds_b128 %0, %1, off"
                 :: "v"(lds_base + (uint32_t)off),
                    "v"((const void*)(src + off))
                 : "memory");
  }
}
__device__ __forceinline__ void wait_async0() {
  asm volatile("s_wait_asynccnt 0" ::: "memory");
}

// ---------------------------------------------------------------------------
// K1: per-head folded matrices (bf16) + bias vectors.  grid = 16 x 256.
// ---------------------------------------------------------------------------
__global__ __launch_bounds__(256) void precompute_kernel(
    const float* __restrict__ W1, const float* __restrict__ b1,
    const float* __restrict__ Wv, const float* __restrict__ bv,
    const float* __restrict__ Wo, const float* __restrict__ bo,
    char* __restrict__ ws) {
  const int h = blockIdx.x;
  const int t = threadIdx.x;
  bf16_t* MTbf = (bf16_t*)(ws + WSB_MTBF);
  bf16_t* CTbf = (bf16_t*)(ws + WSB_CTBF);
  float*  pv   = (float*)(ws + WSB_PV);
  float*  qv   = (float*)(ws + WSB_QV);
  float*  sv   = (float*)(ws + WSB_SV);
  float*  dv   = (float*)(ws + WSB_DV);
  if (h < H_) {
    const int cq = h * DQK_;
    const int ck = H_ * DQK_ + h * DQK_;
    for (int idx = t; idx < 64 * 64; idx += 256) {   // MT[h][j][i]
      const int j = idx >> 6, i = idx & 63;
      float acc = 0.0f;
      for (int d = 0; d < DQK_; ++d)
        acc += W1[i * (2 * H_ * DQK_) + cq + d] * W1[j * (2 * H_ * DQK_) + ck + d];
      MTbf[h * 4096 + idx] = (bf16_t)(acc * RS_);
    }
    for (int idx = t; idx < 64 * 64; idx += 256) {   // CT[h][o][j]
      const int o = idx >> 6, j = idx & 63;
      float acc = 0.0f;
      for (int v = 0; v < DV_; ++v)
        acc += Wv[j * (H_ * DV_) + h * DV_ + v] * Wo[(h * DV_ + v) * DOUT_ + o];
      CTbf[h * 4096 + idx] = (bf16_t)acc;
    }
    if (t < 64) {
      float acc = 0.0f;
      for (int d = 0; d < DQK_; ++d)
        acc += W1[t * (2 * H_ * DQK_) + cq + d] * b1[ck + d];
      pv[h * 64 + t] = acc * RS_;
    } else if (t < 128) {
      const int i = t - 64;
      float acc = 0.0f;
      for (int d = 0; d < DQK_; ++d)
        acc += W1[i * (2 * H_ * DQK_) + ck + d] * b1[cq + d];
      qv[h * 64 + i] = acc * RS_;
    } else if (t == 128) {
      float acc = 0.0f;
      for (int d = 0; d < DQK_; ++d) acc += b1[cq + d] * b1[ck + d];
      sv[h] = acc;
    }
  } else {
    if (t < 64) {
      float acc = bo[t];
      for (int c = 0; c < H_ * DV_; ++c) acc += bv[c] * Wo[c * DOUT_ + t];
      dv[t] = acc;
    }
  }
}

// ---------------------------------------------------------------------------
// K2: stage G as bf16 (both orders) + rank-1 bias projections.
// grid = 128 x 256: blocks 0..7 convert G for batch b; blocks 8..127 compute
// colterm/rowterm for (b,h).  Runs after K1 (needs pv/qv/sv).
// ---------------------------------------------------------------------------
__global__ __launch_bounds__(256) void stage_kernel(
    const float* __restrict__ G, char* __restrict__ ws) {
  const int blk = blockIdx.x;
  const int t = threadIdx.x;
  if (blk < B_) {
    const int b = blk;
    bf16_t* gbf  = (bf16_t*)(ws + WSB_GBF)  + (size_t)b * E_ * DIN_;
    bf16_t* gbfT = (bf16_t*)(ws + WSB_GBFT) + (size_t)b * E_ * DIN_;
    const float* Gb = G + (size_t)b * E_ * DIN_;
    for (int idx = t; idx < E_ * DIN_; idx += 256) {
      const int e = idx >> 6, d = idx & 63;
      const float v = Gb[idx];
      gbf[idx] = (bf16_t)v;
      gbfT[d * 512 + e] = (bf16_t)v;
    }
  } else {
    const int id = blk - B_;
    const int b = id / H_, h = id % H_;
    const float* Gb = G + (size_t)b * E_ * DIN_;
    const float* pv = (const float*)(ws + WSB_PV) + h * 64;
    const float* qv = (const float*)(ws + WSB_QV) + h * 64;
    const float sh  = ((const float*)(ws + WSB_SV))[h];
    float* col = (float*)(ws + WSB_COL) + (size_t)(b * H_ + h) * E_;
    float* row = (float*)(ws + WSB_ROW) + (size_t)(b * H_ + h) * E_;
    for (int k = t; k < E_; k += 256) {
      float ac = sh, ar = 0.0f;
      for (int i = 0; i < DIN_; ++i) {
        const float g = Gb[k * 64 + i];
        ac += g * qv[i];
        ar += g * pv[i];
      }
      col[k] = ac;
      row[k] = ar;
    }
  }
}

// ---------------------------------------------------------------------------
// K3: main attention kernel.  grid = B*H*8 q-tiles, 256 threads (8 waves).
// Wave w: row strip rs = w&3 (16 rows), N-half = w>>2.
// All GEMMs process n-tile PAIRS with a shared A-frag and two independent
// accumulators, so the scheduler can fill WMMA->VALU hazard slots.
// Dynamic LDS (289536 B of 320 KB):
//   [0)      S       f32 [64][512]                     131072
//   [131072) Gbf     bf16 [512][64] (alias AttnBf bf16 [64][512])
//   [196608) GbfT    bf16 [64][512]
//   [262144) MTs     bf16 [64][64]
//   [270336) CTs     bf16 [64][64]
//   [278528) Pbf     bf16 [64][64]  (alias Tbf)
//   [286720) colterm f32 [512]
//   [288768) rowterm f32 [64]
//   [289024) maskk   u8  [512]
// ---------------------------------------------------------------------------
__global__ __launch_bounds__(256) void attn_kernel(
    const int* __restrict__ Gmask, const float* __restrict__ Gbias,
    const char* __restrict__ ws, float* __restrict__ attn_out,
    float* __restrict__ ws_acc) {
  extern __shared__ char smem[];
  float*   S       = (float*)(smem);
  bf16_t*  Gbf     = (bf16_t*)(smem + 131072);
  bf16_t*  AttnBf  = Gbf;                     // reuse after GEMM2 barrier
  bf16_t*  GbfT    = (bf16_t*)(smem + 196608);
  bf16_t*  MTs     = (bf16_t*)(smem + 262144);
  bf16_t*  CTs     = (bf16_t*)(smem + 270336);
  bf16_t*  Pbf     = (bf16_t*)(smem + 278528);
  bf16_t*  Tbf     = Pbf;                     // reuse after GEMM2 barrier
  float*   colterm = (float*)(smem + 286720);
  float*   rowterm = (float*)(smem + 288768);
  uint8_t* maskk   = (uint8_t*)(smem + 289024);
  const uint32_t lds0 = (uint32_t)(uintptr_t)smem;

  const int bid = blockIdx.x;
  const int qt = bid & 7;
  const int h  = (bid >> 3) % H_;
  const int b  = bid / (8 * H_);
  const int tid = threadIdx.x;
  const int lane = tid & 31;
  const int w = tid >> 5;
  const int rs = w & 3;            // row strip (16 rows)
  const int half = w >> 2;         // N-split half
  const int m0 = rs * 16;

  // ---- async-stage bf16 operand tiles into LDS (ASYNCcnt) -----------------
  lds_copy_async(lds0 + 131072, ws + WSB_GBF  + (size_t)b * 65536, 65536, tid);
  lds_copy_async(lds0 + 196608, ws + WSB_GBFT + (size_t)b * 65536, 65536, tid);
  lds_copy_async(lds0 + 262144, ws + WSB_MTBF + (size_t)h * 8192, 8192, tid);
  lds_copy_async(lds0 + 270336, ws + WSB_CTBF + (size_t)h * 8192, 8192, tid);
  lds_copy_async(lds0 + 286720, ws + WSB_COL + (size_t)(b * H_ + h) * 2048,
                 2048, tid);

  // ---- regular staging that overlaps the async copies ---------------------
  if (tid < 64)
    rowterm[tid] =
        ((const float*)(ws + WSB_ROW))[(size_t)(b * H_ + h) * E_ + qt * 64 + tid];
  for (int k = tid; k < E_; k += 256)
    maskk[k] = (Gmask[b * E_ + k] != 0) ? 1 : 0;
  // prefetch this WG's 128 KB G_bias tile (1024 cachelines) into L2/L0
  {
    const char* bias_base =
        (const char*)(Gbias + ((size_t)(b * E_ + qt * 64)) * E_);
    for (int i = tid; i < 1024; i += 256)
      __builtin_prefetch(bias_base + i * 128, 0, 3);
  }
  wait_async0();
  __syncthreads();

  // ---- GEMM1: P = G_qtile(64x64) @ M(64x64); one n-tile pair per wave -----
  {
    const int nt0 = half * 2, nt1 = nt0 + 1;
    f32x8 c0 = zero8(), c1 = zero8();
#pragma unroll
    for (int kc = 0; kc < 2; ++kc) {
      bf16x16 a = load_afrag(Gbf, 64, qt * 64 + m0, kc * 32, lane);
      bf16x16 b0 = load_bfrag(MTs, 64, nt0 * 16, kc * 32, lane);
      bf16x16 b1 = load_bfrag(MTs, 64, nt1 * 16, kc * 32, lane);
      c0 = wmma_bf16(a, b0, c0);
      c1 = wmma_bf16(a, b1, c1);
    }
#pragma unroll
    for (int v = 0; v < 8; ++v) {
      const int m = v + ((lane >> 4) << 3);
      Pbf[(m0 + m) * 64 + nt0 * 16 + (lane & 15)] = (bf16_t)c0[v];
      Pbf[(m0 + m) * 64 + nt1 * 16 + (lane & 15)] = (bf16_t)c1[v];
    }
  }
  __syncthreads();

  // ---- GEMM2: S = P @ G^T + bias + masks; 8 n-tile pairs per wave ---------
  for (int i = 0; i < 16; i += 2) {
    const int nt0 = half * 16 + i, nt1 = nt0 + 1;
    f32x8 c0 = zero8(), c1 = zero8();
#pragma unroll
    for (int kc = 0; kc < 2; ++kc) {
      bf16x16 a = load_afrag(Pbf, 64, m0, kc * 32, lane);
      bf16x16 b0 = load_bfrag(Gbf, 64, nt0 * 16, kc * 32, lane);
      bf16x16 b1 = load_bfrag(Gbf, 64, nt1 * 16, kc * 32, lane);
      c0 = wmma_bf16(a, b0, c0);
      c1 = wmma_bf16(a, b1, c1);
    }
    const int kg0 = nt0 * 16 + (lane & 15);
    const int kg1 = nt1 * 16 + (lane & 15);
#pragma unroll
    for (int v = 0; v < 8; ++v) {
      const int m = v + ((lane >> 4) << 3);
      const int r = m0 + m;
      const int qg = qt * 64 + r;
      const float* brow = Gbias + ((size_t)(b * E_ + qg)) * E_;
      float val0 = c0[v] + rowterm[r] + colterm[kg0] + brow[kg0];
      float val1 = c1[v] + rowterm[r] + colterm[kg1] + brow[kg1];
      if (!maskk[kg0] || (qg == 0 && kg0 >= 2) || (qg >= 2 && kg0 == 0))
        val0 = NEGV;
      if (!maskk[kg1] || (qg == 0 && kg1 >= 2) || (qg >= 2 && kg1 == 0))
        val1 = NEGV;
      S[r * 512 + kg0] = val0;
      S[r * 512 + kg1] = val1;
    }
  }
  __syncthreads();   // Gbf region may now be reused as AttnBf

  // ---- fp32 softmax; wave owns 8 rows; NT-store attn ----------------------
  for (int rr = 0; rr < 8; ++rr) {
    const int r = w * 8 + rr;
    const int qg = qt * 64 + r;
    float* Srow = S + r * 512;
    float vmax = -3.4e38f;
    for (int i = lane; i < E_; i += 32) vmax = fmaxf(vmax, Srow[i]);
#pragma unroll
    for (int off = 16; off; off >>= 1) vmax = fmaxf(vmax, __shfl_xor(vmax, off, 32));
    float ssum = 0.0f;
    for (int i = lane; i < E_; i += 32) {
      const float e = expf(Srow[i] - vmax);   // exp(-1e9 - max) -> exact 0
      Srow[i] = e;
      ssum += e;
    }
#pragma unroll
    for (int off = 16; off; off >>= 1) ssum += __shfl_xor(ssum, off, 32);
    const float inv = 1.0f / ssum;
    float* arow = attn_out + (((size_t)(b * H_ + h) * E_ + qg) * E_);
    for (int i = lane; i < E_; i += 32) {
      const float a = Srow[i] * inv;
      __builtin_nontemporal_store(a, &arow[i]);  // streaming 126 MB output
      AttnBf[r * 512 + i] = (bf16_t)a;
    }
  }
  __syncthreads();

  // ---- GEMM3: T = attn(16x512) @ G(512x64); one n-tile pair per wave ------
  {
    const int nt0 = half * 2, nt1 = nt0 + 1;
    f32x8 c0 = zero8(), c1 = zero8();
#pragma unroll
    for (int kc = 0; kc < 16; ++kc) {
      bf16x16 a = load_afrag(AttnBf, 512, m0, kc * 32, lane);
      bf16x16 b0 = load_bfrag(GbfT, 512, nt0 * 16, kc * 32, lane);
      bf16x16 b1 = load_bfrag(GbfT, 512, nt1 * 16, kc * 32, lane);
      c0 = wmma_bf16(a, b0, c0);
      c1 = wmma_bf16(a, b1, c1);
    }
#pragma unroll
    for (int v = 0; v < 8; ++v) {
      const int m = v + ((lane >> 4) << 3);
      Tbf[(m0 + m) * 64 + nt0 * 16 + (lane & 15)] = (bf16_t)c0[v];
      Tbf[(m0 + m) * 64 + nt1 * 16 + (lane & 15)] = (bf16_t)c1[v];
    }
  }
  __syncthreads();   // GEMM4 A-frags need both column halves of Tbf

  // ---- GEMM4: out_h = T(16x64) @ C(64x64) -> per-head accumulator ---------
  {
    const int nt0 = half * 2, nt1 = nt0 + 1;
    f32x8 c0 = zero8(), c1 = zero8();
#pragma unroll
    for (int kc = 0; kc < 2; ++kc) {
      bf16x16 a = load_afrag(Tbf, 64, m0, kc * 32, lane);
      bf16x16 b0 = load_bfrag(CTs, 64, nt0 * 16, kc * 32, lane);
      bf16x16 b1 = load_bfrag(CTs, 64, nt1 * 16, kc * 32, lane);
      c0 = wmma_bf16(a, b0, c0);
      c1 = wmma_bf16(a, b1, c1);
    }
    float* accp = ws_acc + (size_t)h * (B_ * E_ * DOUT_) +
                  ((size_t)(b * E_) + qt * 64 + m0) * DOUT_;
#pragma unroll
    for (int v = 0; v < 8; ++v) {
      const int m = v + ((lane >> 4) << 3);
      accp[m * DOUT_ + nt0 * 16 + (lane & 15)] = c0[v];
      accp[m * DOUT_ + nt1 * 16 + (lane & 15)] = c1[v];
    }
  }
}

// ---------------------------------------------------------------------------
// K4: deterministic head reduction: out = dv + sum_h acc_h
// ---------------------------------------------------------------------------
__global__ __launch_bounds__(256) void reduce_out_kernel(
    const char* __restrict__ ws, const float* __restrict__ ws_acc,
    float* __restrict__ outp) {
  const int idx = blockIdx.x * 256 + threadIdx.x;   // < 8*512*64
  float acc = ((const float*)(ws + WSB_DV))[idx & 63];
#pragma unroll
  for (int h = 0; h < H_; ++h)
    acc += ws_acc[(size_t)h * (B_ * E_ * DOUT_) + idx];
  outp[idx] = acc;
}

// ---------------------------------------------------------------------------
extern "C" void kernel_launch(void* const* d_in, const int* in_sizes, int n_in,
                              void* d_out, int out_size, void* d_ws, size_t ws_size,
                              hipStream_t stream) {
  (void)in_sizes; (void)n_in; (void)out_size; (void)ws_size;
  const float* G     = (const float*)d_in[0];
  const int*   Gmask = (const int*)  d_in[1];
  const float* Gbias = (const float*)d_in[2];
  const float* W1    = (const float*)d_in[3];
  const float* b1    = (const float*)d_in[4];
  const float* Wv    = (const float*)d_in[5];
  const float* bv    = (const float*)d_in[6];
  const float* Wo    = (const float*)d_in[7];
  const float* bo    = (const float*)d_in[8];

  char*  ws     = (char*)d_ws;
  float* ws_acc = (float*)(ws + WSB_ACC);
  float* attn_p = (float*)d_out;                           // [8,15,512,512]
  float* out_p  = attn_p + (size_t)B_ * H_ * E_ * E_;      // [8,512,64]

  const int smem_bytes = 289536;  // ~283 KB of the 320 KB WGP LDS
  (void)hipFuncSetAttribute((const void*)attn_kernel,
                            hipFuncAttributeMaxDynamicSharedMemorySize,
                            smem_bytes);

  precompute_kernel<<<H_ + 1, 256, 0, stream>>>(W1, b1, Wv, bv, Wo, bo, ws);
  stage_kernel<<<B_ + B_ * H_, 256, 0, stream>>>(G, ws);
  attn_kernel<<<B_ * H_ * 8, 256, smem_bytes, stream>>>(Gmask, Gbias, ws,
                                                        attn_p, ws_acc);
  reduce_out_kernel<<<(B_ * E_ * DOUT_) / 256, 256, 0, stream>>>(ws, ws_acc, out_p);
}
